// DoubleSAGE_49228915147570
// MI455X (gfx1250) — matrix operand order repaired and checked by reading, not compiled, not run on hardware
//
#include <hip/hip_runtime.h>

typedef __attribute__((ext_vector_type(16))) __bf16        v16bf;
typedef __attribute__((ext_vector_type(8)))  float         v8f;
typedef __attribute__((ext_vector_type(8)))  unsigned int  v8u;
typedef __attribute__((ext_vector_type(4)))  int           v4i;

union BFragU { v8u u; v16bf b; };

#if __has_builtin(__builtin_amdgcn_global_load_async_to_lds_b128)
#define ASYNC_LDS 1
typedef v4i __attribute__((address_space(1)))* gas_v4i_ptr;
typedef v4i __attribute__((address_space(3)))* las_v4i_ptr;
#else
#define ASYNC_LDS 0
#endif

#if __has_builtin(__builtin_amdgcn_s_wait_asynccnt)
#define WAIT_ASYNC() __builtin_amdgcn_s_wait_asynccnt(0)
#else
#define WAIT_ASYNC() asm volatile("s_wait_asynccnt 0" ::: "memory")
#endif

__device__ __forceinline__ unsigned short f2bf(float f) {
  unsigned int u = __float_as_uint(f);
  u += 0x7FFFu + ((u >> 16) & 1u);   // round-to-nearest-even
  return (unsigned short)(u >> 16);
}

__device__ __forceinline__ BFragU frag_from(uint4 lo, uint4 hi) {
  BFragU r;
  r.u = (v8u){lo.x, lo.y, lo.z, lo.w, hi.x, hi.y, hi.z, hi.w};
  return r;
}

// ---------------- scatter: agg[dst] += feat[src]; deg[dst] += 1 ----------------
__global__ __launch_bounds__(256) void scatter_add_kernel(
    const float* __restrict__ feat, const int* __restrict__ src,
    const int* __restrict__ dst, float* __restrict__ agg,
    float* __restrict__ deg, int E, int d) {
  int lane = threadIdx.x & 31;
  int wave = threadIdx.x >> 5;          // wave32
  int e = blockIdx.x * 8 + wave;
  if (e >= E) return;
  int s = src[e], t = dst[e];
  const float* fr = feat + (size_t)s * d;
  float*       ar = agg  + (size_t)t * d;
  __builtin_prefetch(fr, 0, 0);         // global_prefetch_b8
  for (int j = lane; j < d; j += 32)
    atomicAdd(&ar[j], fr[j]);
  if (deg != nullptr && lane == 0)
    atomicAdd(&deg[t], 1.0f);
}

__global__ __launch_bounds__(256) void invdeg_kernel(
    const float* __restrict__ deg, float* __restrict__ invd, int n) {
  int i = blockIdx.x * blockDim.x + threadIdx.x;
  if (i < n) invd[i] = 1.0f / fmaxf(deg[i], 1.0f);
}

// -------- f32 -> bf16 conversion (optionally row-scaled by invd), 8 elems/thread ----
__global__ __launch_bounds__(256) void cvt_bf16_kernel(
    const float* __restrict__ in, const float* __restrict__ invd, int d,
    unsigned short* __restrict__ out, long long n) {
  long long base = ((long long)blockIdx.x * blockDim.x + threadIdx.x) * 8;
  if (base >= n) return;
  float scale = (invd != nullptr) ? invd[base / d] : 1.0f;   // d % 8 == 0
  const float* p = in + base;
  uint4 v;
  v.x = (unsigned)f2bf(p[0] * scale) | ((unsigned)f2bf(p[1] * scale) << 16);
  v.y = (unsigned)f2bf(p[2] * scale) | ((unsigned)f2bf(p[3] * scale) << 16);
  v.z = (unsigned)f2bf(p[4] * scale) | ((unsigned)f2bf(p[5] * scale) << 16);
  v.w = (unsigned)f2bf(p[6] * scale) | ((unsigned)f2bf(p[7] * scale) << 16);
  *(uint4*)(out + base) = v;
}

// -------- repack weight [K x N] f32 -> bf16 in WMMA B-fragment order --------------
// Wr layout: [kt=K/32][ng=N/16][lane=32][i=16]; lane: h=lane>>4, nloc=lane&15;
// element i of lane = W[(kt*32 + h*16 + i) * N + ng*16 + nloc]
__global__ __launch_bounds__(256) void repack_w_kernel(
    const float* __restrict__ W, unsigned short* __restrict__ Wr, int K, int N) {
  int o = blockIdx.x * blockDim.x + threadIdx.x;
  if (o >= K * N) return;
  int i    = o & 15;
  int lane = (o >> 4) & 31;
  int t2   = o >> 9;
  int ngn  = N >> 4;
  int ng   = t2 % ngn;
  int kt   = t2 / ngn;
  int h    = lane >> 4, nloc = lane & 15;
  int k = kt * 32 + h * 16 + i;
  int n = ng * 16 + nloc;
  Wr[o] = f2bf(W[(size_t)k * N + n]);
}

// ---------------- fused SAGE GEMM (bf16 operands, f32 accumulate) ------------------
// out = act(Aself*Wself + Aneigh*Wneigh + bias).  Block: 256 thr (8 waves);
// tile 128(M) x 64(N); each wave: 32(M) x 32(N) = 4 WMMAs per K-step.
__global__ __launch_bounds__(256) void sage_gemm_bf16_kernel(
    const unsigned short* __restrict__ AbfS, const unsigned short* __restrict__ AbfN,
    const unsigned short* __restrict__ WrS,  const unsigned short* __restrict__ WrN,
    const float* __restrict__ bias, float* __restrict__ out,
    int M, int K, int Nc, int relu) {
  __shared__ __align__(16) unsigned int Afrag[8][32][8];   // 8 m-groups x 32 lanes x 32B
  int tid  = threadIdx.x;
  int lane = tid & 31, wave = tid >> 5;
  int wm = wave >> 1, wn = wave & 1;
  int hh = lane >> 4, l15 = lane & 15;
  int bm = blockIdx.x * 128, bn = blockIdx.y * 64;
  int ngn = Nc >> 4;
  bool fullM = (bm + 128 <= M);          // block-uniform: async fast path legal

  // A loader: two 16B half-fragments per thread (f and f+256 of 512)
  int g0 = tid >> 6,           g1 = (tid >> 6) + 4;
  int r6 = tid & 63;
  int lane_ld = r6 >> 1;
  int half    = r6 & 1;
  int mrow_ld = (lane_ld & 15);
  int m0_ld = bm + g0 * 16 + mrow_ld;
  int m1_ld = bm + g1 * 16 + mrow_ld;
  int k_ld  = (lane_ld >> 4) * 16 + half * 8;   // fragment-run start within 32-K tile

  v8f acc00 = {}, acc01 = {}, acc10 = {}, acc11 = {};
  int ng0 = (bn >> 4) + wn * 2, ng1 = ng0 + 1;

  for (int s = 0; s < 2; ++s) {
    const unsigned short* A  = (s == 0) ? AbfS : AbfN;
    const unsigned short* Wr = (s == 0) ? WrS  : WrN;
    for (int kb = 0; kb < K; kb += 32) {
#if ASYNC_LDS
      if (fullM) {
        // direct global -> LDS, no VGPR round trip (ASYNCcnt-tracked)
        __builtin_amdgcn_global_load_async_to_lds_b128(
            (gas_v4i_ptr)(A + (size_t)m0_ld * K + kb + k_ld),
            (las_v4i_ptr)&Afrag[g0][lane_ld][half * 4], 0, 0);
        __builtin_amdgcn_global_load_async_to_lds_b128(
            (gas_v4i_ptr)(A + (size_t)m1_ld * K + kb + k_ld),
            (las_v4i_ptr)&Afrag[g1][lane_ld][half * 4], 0, 0);
        WAIT_ASYNC();
      } else
#endif
      {
        uint4 a0, a1;
        if (m0_ld < M) a0 = *(const uint4*)(A + (size_t)m0_ld * K + kb + k_ld);
        else           a0 = make_uint4(0u, 0u, 0u, 0u);
        if (m1_ld < M) a1 = *(const uint4*)(A + (size_t)m1_ld * K + kb + k_ld);
        else           a1 = make_uint4(0u, 0u, 0u, 0u);
        *(uint4*)&Afrag[g0][lane_ld][half * 4] = a0;
        *(uint4*)&Afrag[g1][lane_ld][half * 4] = a1;
      }
      __syncthreads();

      uint4 a0lo = *(const uint4*)&Afrag[2 * wm][lane][0];
      uint4 a0hi = *(const uint4*)&Afrag[2 * wm][lane][4];
      uint4 a1lo = *(const uint4*)&Afrag[2 * wm + 1][lane][0];
      uint4 a1hi = *(const uint4*)&Afrag[2 * wm + 1][lane][4];
      BFragU fa0 = frag_from(a0lo, a0hi);
      BFragU fa1 = frag_from(a1lo, a1hi);

      int kt = kb >> 5;
      const uint4* bp0 = (const uint4*)(Wr + (((size_t)kt * ngn + ng0) * 32 + lane) * 16);
      const uint4* bp1 = (const uint4*)(Wr + (((size_t)kt * ngn + ng1) * 32 + lane) * 16);
      BFragU fb0 = frag_from(bp0[0], bp0[1]);
      BFragU fb1 = frag_from(bp1[0], bp1[1]);

      acc00 = __builtin_amdgcn_wmma_f32_16x16x32_bf16(false, fa0.b, false, fb0.b, (short)0, acc00, false, false);
      acc01 = __builtin_amdgcn_wmma_f32_16x16x32_bf16(false, fa0.b, false, fb1.b, (short)0, acc01, false, false);
      acc10 = __builtin_amdgcn_wmma_f32_16x16x32_bf16(false, fa1.b, false, fb0.b, (short)0, acc10, false, false);
      acc11 = __builtin_amdgcn_wmma_f32_16x16x32_bf16(false, fa1.b, false, fb1.b, (short)0, acc11, false, false);
      __syncthreads();
    }
  }

  int n0 = bn + wn * 32 + l15, n1 = n0 + 16;
  float bb0 = bias[n0], bb1 = bias[n1];
  for (int j = 0; j < 8; ++j) {          // C layout: VGPR j -> M = hh*8 + j
    int ma = bm + (2 * wm) * 16 + hh * 8 + j;
    int mb = bm + (2 * wm + 1) * 16 + hh * 8 + j;
    if (ma < M) {
      float v0 = acc00[j] + bb0, v1 = acc01[j] + bb1;
      if (relu) { v0 = fmaxf(v0, 0.0f); v1 = fmaxf(v1, 0.0f); }
      out[(size_t)ma * Nc + n0] = v0;
      out[(size_t)ma * Nc + n1] = v1;
    }
    if (mb < M) {
      float v0 = acc10[j] + bb0, v1 = acc11[j] + bb1;
      if (relu) { v0 = fmaxf(v0, 0.0f); v1 = fmaxf(v1, 0.0f); }
      out[(size_t)mb * Nc + n0] = v0;
      out[(size_t)mb * Nc + n1] = v1;
    }
  }
}

// ---------------- edge MLP: score = Wp2 . relu([h_u;h_v] Wp1 + bp1) + bp2 ----------
__global__ __launch_bounds__(256) void edge_mlp_kernel(
    const unsigned short* __restrict__ h2bf, const int* __restrict__ src,
    const int* __restrict__ dst,
    const unsigned short* __restrict__ Wrp1, const float* __restrict__ bp1,
    const float* __restrict__ Wp2, const float* __restrict__ bp2,
    float* __restrict__ out, int EP) {
  __shared__ __align__(16) unsigned int Afrag[4][4][32][8];  // [kt][g][lane][8 uints]
  __shared__ float sc[64];
  int tid  = threadIdx.x;
  int lane = tid & 31, wave = tid >> 5;
  int wm = wave >> 1, wn = wave & 1;
  int hh = lane >> 4, l15 = lane & 15;
  int bm = blockIdx.x * 64;

  for (int q = 0; q < 4; ++q) {          // gather [h2[src];h2[dst]] in fragment order
    int f       = q * 256 + tid;
    int kt      = f >> 8;
    int rem     = f & 255;
    int gg      = rem >> 6;
    int r6      = rem & 63;
    int lane_ld = r6 >> 1;
    int half    = r6 & 1;
    int e  = bm + gg * 16 + (lane_ld & 15);
    int c0 = kt * 32 + (lane_ld >> 4) * 16 + half * 8;
    uint4 v;
    if (e < EP) {
      const unsigned short* hp = (c0 < 64)
          ? h2bf + (size_t)src[e] * 64 + c0
          : h2bf + (size_t)dst[e] * 64 + (c0 - 64);
      v = *(const uint4*)hp;
    } else {
      v = make_uint4(0u, 0u, 0u, 0u);
    }
    *(uint4*)&Afrag[kt][gg][lane_ld][half * 4] = v;
  }
  if (tid < 64) sc[tid] = 0.0f;
  __syncthreads();

  v8f acc0 = {}, acc1 = {};
  int ng0 = wn * 2, ng1 = ng0 + 1;
  for (int kt = 0; kt < 4; ++kt) {
    uint4 alo = *(const uint4*)&Afrag[kt][wm][lane][0];
    uint4 ahi = *(const uint4*)&Afrag[kt][wm][lane][4];
    BFragU fa = frag_from(alo, ahi);
    const uint4* bp0 = (const uint4*)(Wrp1 + (((size_t)kt * 4 + ng0) * 32 + lane) * 16);
    const uint4* bp1 = (const uint4*)(Wrp1 + (((size_t)kt * 4 + ng1) * 32 + lane) * 16);
    BFragU fb0 = frag_from(bp0[0], bp0[1]);
    BFragU fb1 = frag_from(bp1[0], bp1[1]);
    acc0 = __builtin_amdgcn_wmma_f32_16x16x32_bf16(false, fa.b, false, fb0.b, (short)0, acc0, false, false);
    acc1 = __builtin_amdgcn_wmma_f32_16x16x32_bf16(false, fa.b, false, fb1.b, (short)0, acc1, false, false);
  }

  int n0 = wn * 32 + l15, n1 = n0 + 16;
  float w20 = Wp2[n0], w21 = Wp2[n1];
  float bb0 = bp1[n0], bb1 = bp1[n1];
  for (int j = 0; j < 8; ++j) {
    int m = wm * 16 + hh * 8 + j;
    float v0 = fmaxf(acc0[j] + bb0, 0.0f);
    float v1 = fmaxf(acc1[j] + bb1, 0.0f);
    atomicAdd(&sc[m], v0 * w20 + v1 * w21);  // ds_add_f32 reduction over N
  }
  __syncthreads();
  if (tid < 64) {
    int e = bm + tid;
    if (e < EP) out[e] = sc[tid] + bp2[0];
  }
}

extern "C" void kernel_launch(void* const* d_in, const int* in_sizes, int n_in,
                              void* d_out, int out_size, void* d_ws, size_t ws_size,
                              hipStream_t stream) {
  const float* x       = (const float*)d_in[0];
  const int*   esrc    = (const int*)d_in[1];
  const int*   edst    = (const int*)d_in[2];
  const int*   psrc    = (const int*)d_in[3];
  const int*   pdst    = (const int*)d_in[4];
  const int*   gsrc    = (const int*)d_in[5];
  const int*   gdst    = (const int*)d_in[6];
  const float* Wself1  = (const float*)d_in[7];
  const float* Wneigh1 = (const float*)d_in[8];
  const float* b1      = (const float*)d_in[9];
  const float* Wself2  = (const float*)d_in[10];
  const float* Wneigh2 = (const float*)d_in[11];
  const float* b2      = (const float*)d_in[12];
  const float* Wp1     = (const float*)d_in[13];
  const float* bp1     = (const float*)d_in[14];
  const float* Wp2     = (const float*)d_in[15];
  const float* bp2     = (const float*)d_in[16];

  const int N  = in_sizes[0] / 512;
  const int E  = in_sizes[1];
  const int EP = in_sizes[3];

  // Workspace layout (f32 slot units); bf16 buffers use 2 elems per slot.
  float* fw   = (float*)d_ws;
  size_t off  = 0;
  float* aggF = fw + off; off += (size_t)N * 512;   // agg1 f32; agg2 aliases
  float* deg  = fw + off; off += N;
  float* invd = fw + off; off += N;
  unsigned short* bfA = (unsigned short*)(fw + off); off += (size_t)N * 256; // xbf / h1bf / h2bf
  unsigned short* bfB = (unsigned short*)(fw + off); off += (size_t)N * 256; // agg1bf / agg2bf
  float* hF   = fw + off; off += (size_t)N * 256;   // h1 f32; h2 f32 aliases
  unsigned short* Wr1  = (unsigned short*)(fw + off); off += (512 * 256 * 2) / 2;
  unsigned short* Wr2  = (unsigned short*)(fw + off); off += (256 * 64 * 2) / 2;
  unsigned short* Wrp1 = (unsigned short*)(fw + off); off += (128 * 64) / 2;

  const long long nx  = (long long)N * 512;
  const long long nh1 = (long long)N * 256;
  const long long nh2 = (long long)N * 64;

  // ---- Layer 1 ----
  (void)hipMemsetAsync(aggF, 0, (size_t)N * 512 * sizeof(float), stream);
  (void)hipMemsetAsync(deg,  0, (size_t)N * sizeof(float), stream);
  scatter_add_kernel<<<(E + 7) / 8, 256, 0, stream>>>(x, esrc, edst, aggF, deg, E, 512);
  invdeg_kernel<<<(N + 255) / 256, 256, 0, stream>>>(deg, invd, N);
  cvt_bf16_kernel<<<(int)((nx / 8 + 255) / 256), 256, 0, stream>>>(x, nullptr, 512, bfA, nx);
  cvt_bf16_kernel<<<(int)((nx / 8 + 255) / 256), 256, 0, stream>>>(aggF, invd, 512, bfB, nx);
  repack_w_kernel<<<(512 * 256 + 255) / 256, 256, 0, stream>>>(Wself1,  Wr1,             512, 256);
  repack_w_kernel<<<(512 * 256 + 255) / 256, 256, 0, stream>>>(Wneigh1, Wr1 + 512 * 256, 512, 256);
  dim3 g1((N + 127) / 128, 256 / 64);
  sage_gemm_bf16_kernel<<<g1, 256, 0, stream>>>(bfA, bfB, Wr1, Wr1 + 512 * 256, b1, hF,
                                                N, 512, 256, 1);
  // ---- Layer 2 (agg f32 region reused) ----
  (void)hipMemsetAsync(aggF, 0, (size_t)N * 256 * sizeof(float), stream);
  scatter_add_kernel<<<(E + 7) / 8, 256, 0, stream>>>(hF, esrc, edst, aggF, nullptr, E, 256);
  cvt_bf16_kernel<<<(int)((nh1 / 8 + 255) / 256), 256, 0, stream>>>(hF, nullptr, 256, bfA, nh1);
  cvt_bf16_kernel<<<(int)((nh1 / 8 + 255) / 256), 256, 0, stream>>>(aggF, invd, 256, bfB, nh1);
  repack_w_kernel<<<(256 * 64 + 255) / 256, 256, 0, stream>>>(Wself2,  Wr2,            256, 64);
  repack_w_kernel<<<(256 * 64 + 255) / 256, 256, 0, stream>>>(Wneigh2, Wr2 + 256 * 64, 256, 64);
  float* h2F = hF;   // h1 f32 dead after conversion; GEMM2 reads bf16 copies only
  dim3 g2((N + 127) / 128, 1);
  sage_gemm_bf16_kernel<<<g2, 256, 0, stream>>>(bfA, bfB, Wr2, Wr2 + 256 * 64, b2, h2F,
                                                N, 256, 64, 0);
  // ---- Edge MLP ----
  unsigned short* h2bf = bfA;   // h1bf dead after GEMM2
  cvt_bf16_kernel<<<(int)((nh2 / 8 + 255) / 256), 256, 0, stream>>>(h2F, nullptr, 64, h2bf, nh2);
  repack_w_kernel<<<(128 * 64 + 255) / 256, 256, 0, stream>>>(Wp1, Wrp1, 128, 64);
  edge_mlp_kernel<<<(EP + 63) / 64, 256, 0, stream>>>(h2bf, psrc, pdst, Wrp1, bp1, Wp2, bp2,
                                                      (float*)d_out, EP);
  edge_mlp_kernel<<<(EP + 63) / 64, 256, 0, stream>>>(h2bf, gsrc, gdst, Wrp1, bp1, Wp2, bp2,
                                                      (float*)d_out + EP, EP);
}
